// GMN_match_hinge_vaibhav_62886911148483
// MI455X (gfx1250) — matrix operand order repaired and checked by reading, not compiled, not run on hardware
//
#include <hip/hip_runtime.h>
#include <hip/hip_bf16.h>

// ---------------- problem constants ----------------
constexpr int kB   = 256;            // batch pairs
constexpr int kN   = 96;             // nodes per graph
constexpr int kMAXN= 128;
constexpr int kD   = 128;            // hidden dim
constexpr int kIN  = 128;
constexpr int kEIN = 32;
constexpr int kEPG = 480;
constexpr int kNG  = 2 * kB;                 // 512 graphs
constexpr int kNODES = kNG * kN;             // 49152
constexpr int kEDGES = kNG * kEPG;           // 245760
constexpr int kSKIT  = 20;
constexpr float kInvTemp = 10.0f;            // 1/SK_TEMP

// ---------------- WMMA types / helpers ----------------
typedef __bf16 bf16_t;
typedef __attribute__((ext_vector_type(16))) __bf16 v16bf;
typedef __attribute__((ext_vector_type(8)))  float  v8f;

#define DEV static __device__ __forceinline__

DEV v8f wmma_bf16(v16bf a, v16bf b, v8f c) {
  // D = A(16x32 bf16) * B(32x16 bf16) + C(16x16 f32)
  return __builtin_amdgcn_wmma_f32_16x16x32_bf16(false, a, false, b,
                                                 (short)0, c, false, false);
}

// ---- async global->LDS 16B copy (ASYNCcnt path, CDNA5 ISA) ----
// LDS dest address = low 32 bits of the flat LDS pointer (ISA 10.2 aperture rule).
DEV void asy_cp16(const bf16_t* g, bf16_t* l) {
  uint32_t ldsOff = (uint32_t)(uintptr_t)l;
  asm volatile("global_load_async_to_lds_b128 %0, %1, off"
               :: "v"(ldsOff), "v"((uint64_t)(uintptr_t)g)
               : "memory");
}
DEV void asy_wait0() { asm volatile("s_wait_asynccnt 0" ::: "memory"); }

// ---- packed bf16 fragment loaders (no converts; (k,k+1) pairs contiguous) ----

// A fragment from row-major bf16: A[m][k] = S[(m0+m)*ld + k0+k]
DEV v16bf ldA_bf(const bf16_t* S, int ld, int m0, int k0, int lane) {
  int m  = m0 + (lane & 15);
  int kb = (lane < 16) ? 0 : 16;
  const uint32_t* p = (const uint32_t*)(S + (size_t)m * ld + k0 + kb);
  union { v16bf v; uint32_t u[8]; } r;
#pragma unroll
  for (int i = 0; i < 8; ++i) r.u[i] = p[i];  // pairs 0..7
  return r.v;
}

// B fragment from K-contiguous bf16 (W^T stored as [n][k], ld = K)
DEV v16bf ldB_bfT(const bf16_t* S, int ld, int k0, int n0, int lane) {
  int n  = n0 + (lane & 15);
  int kb = (lane < 16) ? 0 : 16;
  const uint32_t* p = (const uint32_t*)(S + (size_t)n * ld + k0 + kb);
  union { v16bf v; uint32_t u[8]; } r;
#pragma unroll
  for (int j = 0; j < 8; ++j) r.u[j] = p[j];                     // pairs 0..7
  return r.v;
}

// ---- f32-source loaders (convert on read; only the tiny Sinkhorn GEMM) ----
DEV v16bf ldA_rm_f32(const float* S, int ld, int m0, int k0, int lane) {
  int m  = m0 + (lane & 15);
  int kb = (lane < 16) ? 0 : 16;
  v16bf a;
#pragma unroll
  for (int i = 0; i < 8; ++i) {
    int k = k0 + kb + 2 * i;
    a[2 * i]     = (__bf16)S[m * ld + k];
    a[2 * i + 1] = (__bf16)S[m * ld + k + 1];
  }
  return a;
}
DEV v16bf ldB_rm_bound_f32(const float* S, int ld, int k0, int n0, int lane, int kmax) {
  int n  = n0 + (lane & 15);
  int kb = (lane < 16) ? 0 : 16;
  v16bf b;
#pragma unroll
  for (int j = 0; j < 8; ++j) {
    int k = k0 + kb + 2 * j;
    b[2 * j]     = (__bf16)((k     < kmax) ? S[(size_t)k * ld + n]       : 0.0f);
    b[2 * j + 1] = (__bf16)((k + 1 < kmax) ? S[(size_t)(k + 1) * ld + n] : 0.0f);
  }
  return b;
}

// ---------------- kernels ----------------

__global__ void k_zero(float* __restrict__ p, size_t n) {
  size_t i = (size_t)blockIdx.x * blockDim.x + threadIdx.x;
  size_t st = (size_t)gridDim.x * blockDim.x;
  for (; i < n; i += st) p[i] = 0.0f;
}

// W[K][N] f32  ->  Wt[N][K] bf16
__global__ void k_wprep(const float* __restrict__ W, bf16_t* __restrict__ Wt, int K, int N) {
  int i = blockIdx.x * 256 + threadIdx.x;
  if (i < K * N) {
    int k = i / N, n = i - k * N;
    Wt[(size_t)n * K + k] = (bf16_t)W[i];
  }
}

// h = X @ W + b ; writes f32 master + bf16 shadow
__global__ void k_encode_nodes(const float* __restrict__ X, const bf16_t* __restrict__ Wt,
                               const float* __restrict__ bias, float* __restrict__ H,
                               bf16_t* __restrict__ HB) {
  __shared__ alignas(16) bf16_t sA[16 * kIN];
  int row0 = blockIdx.x * 16;
  int tid = threadIdx.x;
  for (int i = tid; i < 16 * kIN; i += 256) sA[i] = (bf16_t)X[(size_t)row0 * kIN + i];
  __syncthreads();
  int wave = tid >> 5, lane = tid & 31;
  int n0 = wave * 16;
  v8f c = {};
#pragma unroll
  for (int kk = 0; kk < kIN; kk += 32)
    c = wmma_bf16(ldA_bf(sA, kIN, 0, kk, lane), ldB_bfT(Wt, kIN, kk, n0, lane), c);
  int n = n0 + (lane & 15), mB = (lane < 16) ? 0 : 8;
  float bv = bias[n];
#pragma unroll
  for (int r = 0; r < 8; ++r) {
    float v = c[r] + bv;
    size_t idx = (size_t)(row0 + mB + r) * kD + n;
    H[idx] = v;
    HB[idx] = (bf16_t)v;
  }
}

// e = EF @ W + b  (bf16-only output; e is only ever a WMMA operand)
__global__ void k_encode_edges(const float* __restrict__ X, const bf16_t* __restrict__ Wt,
                               const float* __restrict__ bias, bf16_t* __restrict__ EB) {
  __shared__ alignas(16) bf16_t sA[16 * kEIN];
  int row0 = blockIdx.x * 16;
  int tid = threadIdx.x;
  for (int i = tid; i < 16 * kEIN; i += 256) sA[i] = (bf16_t)X[(size_t)row0 * kEIN + i];
  __syncthreads();
  int wave = tid >> 5, lane = tid & 31;
  int n0 = wave * 16;
  v8f c = {};
  c = wmma_bf16(ldA_bf(sA, kEIN, 0, 0, lane), ldB_bfT(Wt, kEIN, 0, n0, lane), c);
  int n = n0 + (lane & 15), mB = (lane < 16) ? 0 : 8;
  float bv = bias[n];
#pragma unroll
  for (int r = 0; r < 8; ++r)
    EB[(size_t)(row0 + mB + r) * kD + n] = (bf16_t)(c[r] + bv);
}

// fused message MLP + segment-sum: 16 edges/block; gather staged via ASYNC DMA
__global__ void k_msg(const bf16_t* __restrict__ HB, const bf16_t* __restrict__ EB,
                      const int* __restrict__ from_idx, const int* __restrict__ to_idx,
                      const bf16_t* __restrict__ W1t, const float* __restrict__ b1,
                      const bf16_t* __restrict__ W2t, const float* __restrict__ b2,
                      float* __restrict__ AGG) {
  __shared__ alignas(16) bf16_t sIn[16 * 384];
  __shared__ alignas(16) bf16_t sHid[16 * 256];
  int e0 = blockIdx.x * 16;
  int tid = threadIdx.x;
  // 768 x 16B units; per row r: units 0-15 = h[from], 16-31 = h[to], 32-47 = e
  for (int u = tid; u < 768; u += 256) {
    int r = u / 48, part = u - r * 48;
    int eidx = e0 + r;
    const bf16_t* src;
    if (part < 16)      src = HB + (size_t)from_idx[eidx] * kD + part * 8;
    else if (part < 32) src = HB + (size_t)to_idx[eidx] * kD + (part - 16) * 8;
    else                src = EB + (size_t)eidx * kD + (part - 32) * 8;
    asy_cp16(src, sIn + u * 8);
  }
  asy_wait0();
  __syncthreads();
  int wave = tid >> 5, lane = tid & 31;
#pragma unroll
  for (int t = 0; t < 2; ++t) {               // layer 1: 16 col-tiles, 2 per wave
    int n0 = (wave + t * 8) * 16;
    v8f c = {};
#pragma unroll
    for (int kk = 0; kk < 384; kk += 32)
      c = wmma_bf16(ldA_bf(sIn, 384, 0, kk, lane), ldB_bfT(W1t, 384, kk, n0, lane), c);
    int n = n0 + (lane & 15), mB = (lane < 16) ? 0 : 8;
    float bv = b1[n];
#pragma unroll
    for (int r = 0; r < 8; ++r) {
      float v = c[r] + bv;
      sHid[(mB + r) * 256 + n] = (bf16_t)(v > 0.0f ? v : 0.0f);
    }
  }
  __syncthreads();
#pragma unroll
  for (int t = 0; t < 2; ++t) {               // layer 2 + atomic aggregate
    int n0 = (wave + t * 8) * 16;
    v8f c = {};
#pragma unroll
    for (int kk = 0; kk < 256; kk += 32)
      c = wmma_bf16(ldA_bf(sHid, 256, 0, kk, lane), ldB_bfT(W2t, 256, kk, n0, lane), c);
    int n = n0 + (lane & 15), mB = (lane < 16) ? 0 : 8;
    float bv = b2[n];
#pragma unroll
    for (int r = 0; r < 8; ++r) {
      int eidx = e0 + mB + r;
      unsafeAtomicAdd(&AGG[(size_t)to_idx[eidx] * 256 + n], c[r] + bv);
    }
  }
}

// per-pair cross attention, fully in LDS (bf16 operands, ~168KB dynamic)
__global__ void k_cross(const float* __restrict__ H, const bf16_t* __restrict__ HBg,
                        float* __restrict__ CR) {
  extern __shared__ float smf[];
  float*  sS   = smf;                                   // 96*96 f32 sim
  bf16_t* sQ   = (bf16_t*)(smf + 96 * 96);              // 96x128  (row-major)
  bf16_t* sC   = sQ  + 96 * 128;                        // 96x128
  bf16_t* sQt  = sC  + 96 * 128;                        // 128x96  (transposed)
  bf16_t* sCt  = sQt + 128 * 96;                        // 128x96
  bf16_t* sPq  = sCt + 128 * 96;                        // 96x96   (row softmax)
  bf16_t* sPcT = sPq + 96 * 96;                         // 96x96   (col softmax, transposed)
  int b = blockIdx.x;
  const float*  Q  = H   + (size_t)(2 * b) * kN * kD;
  const float*  Cm = H   + (size_t)(2 * b + 1) * kN * kD;
  const bf16_t* Qb = HBg + (size_t)(2 * b) * kN * kD;
  const bf16_t* Cb = HBg + (size_t)(2 * b + 1) * kN * kD;
  int tid = threadIdx.x;
  for (int i = tid; i < kN * kD; i += 256) {
    int r = i >> 7, c = i & 127;
    bf16_t qv = Qb[i], cv = Cb[i];
    sQ[i] = qv;  sQt[c * 96 + r] = qv;
    sC[i] = cv;  sCt[c * 96 + r] = cv;
  }
  __syncthreads();
  int wave = tid >> 5, lane = tid & 31;
  // sim = Q @ C^T  (96x96, 36 tiles); B[k][n] = C[n][k] -> K-contiguous in sC
  for (int t = wave; t < 36; t += 8) {
    int m0 = (t / 6) * 16, n0 = (t % 6) * 16;
    v8f c = {};
#pragma unroll
    for (int kk = 0; kk < kD; kk += 32)
      c = wmma_bf16(ldA_bf(sQ, kD, m0, kk, lane), ldB_bfT(sC, kD, kk, n0, lane), c);
    int n = n0 + (lane & 15), mB = (lane < 16) ? 0 : 8;
#pragma unroll
    for (int r = 0; r < 8; ++r) sS[(m0 + mB + r) * 96 + n] = c[r];
  }
  __syncthreads();
  // row softmax -> sPq (row-major), col softmax -> sPcT (transposed)
  for (int r = tid; r < 96; r += 256) {
    float mx = -3.4e38f;
    for (int j = 0; j < 96; ++j) mx = fmaxf(mx, sS[r * 96 + j]);
    float s = 0.0f;
    float ev[96];
    for (int j = 0; j < 96; ++j) { ev[j] = __expf(sS[r * 96 + j] - mx); s += ev[j]; }
    float inv = 1.0f / s;
    for (int j = 0; j < 96; ++j) sPq[r * 96 + j] = (bf16_t)(ev[j] * inv);
  }
  for (int cI = tid; cI < 96; cI += 256) {
    float mx = -3.4e38f;
    for (int i = 0; i < 96; ++i) mx = fmaxf(mx, sS[i * 96 + cI]);
    float s = 0.0f;
    float ev[96];
    for (int i = 0; i < 96; ++i) { ev[i] = __expf(sS[i * 96 + cI] - mx); s += ev[i]; }
    float inv = 1.0f / s;
    for (int i = 0; i < 96; ++i) sPcT[cI * 96 + i] = (bf16_t)(ev[i] * inv);
  }
  __syncthreads();
  float* CRq = CR + (size_t)(2 * b) * kN * kD;
  float* CRc = CR + (size_t)(2 * b + 1) * kN * kD;
  // cross_q = q - Pq @ C ; B K-contiguous in sCt ([d][node])
  for (int t = wave; t < 48; t += 8) {
    int m0 = (t / 8) * 16, n0 = (t % 8) * 16;
    v8f c = {};
#pragma unroll
    for (int kk = 0; kk < 96; kk += 32)
      c = wmma_bf16(ldA_bf(sPq, 96, m0, kk, lane), ldB_bfT(sCt, 96, kk, n0, lane), c);
    int n = n0 + (lane & 15), mB = (lane < 16) ? 0 : 8;
#pragma unroll
    for (int r = 0; r < 8; ++r) {
      int m = m0 + mB + r;
      CRq[(size_t)m * kD + n] = Q[m * kD + n] - c[r];
    }
  }
  // cross_c = c - Pc^T @ Q ; A rows already transposed, B from sQt
  for (int t = wave; t < 48; t += 8) {
    int m0 = (t / 8) * 16, n0 = (t % 8) * 16;
    v8f c = {};
#pragma unroll
    for (int kk = 0; kk < 96; kk += 32)
      c = wmma_bf16(ldA_bf(sPcT, 96, m0, kk, lane), ldB_bfT(sQt, 96, kk, n0, lane), c);
    int n = n0 + (lane & 15), mB = (lane < 16) ? 0 : 8;
#pragma unroll
    for (int r = 0; r < 8; ++r) {
      int m = m0 + mB + r;
      CRc[(size_t)m * kD + n] = Cm[m * kD + n] - c[r];
    }
  }
}

// fused update MLP: h' = relu([h|agg|cross] @ W1 + b1) @ W2 + b2 ; 16 nodes/block
__global__ void k_update(const bf16_t* __restrict__ HB, const float* __restrict__ AGG,
                         const float* __restrict__ CR,
                         const bf16_t* __restrict__ W1t, const float* __restrict__ b1,
                         const bf16_t* __restrict__ W2t, const float* __restrict__ b2,
                         float* __restrict__ Hn, bf16_t* __restrict__ HBn) {
  __shared__ alignas(16) bf16_t sIn[16 * 512];
  __shared__ alignas(16) bf16_t sHid[16 * 256];
  int row0 = blockIdx.x * 16;
  int tid = threadIdx.x;
  for (int i = tid; i < 16 * 512; i += 256) {
    int r = i >> 9, col = i & 511;
    int node = row0 + r;
    bf16_t v;
    if (col < 128)       v = HB[(size_t)node * kD + col];
    else if (col < 384)  v = (bf16_t)AGG[(size_t)node * 256 + (col - 128)];
    else                 v = (bf16_t)CR[(size_t)node * kD + (col - 384)];
    sIn[i] = v;
  }
  __syncthreads();
  int wave = tid >> 5, lane = tid & 31;
#pragma unroll
  for (int t = 0; t < 2; ++t) {
    int n0 = (wave + t * 8) * 16;
    v8f c = {};
#pragma unroll
    for (int kk = 0; kk < 512; kk += 32)
      c = wmma_bf16(ldA_bf(sIn, 512, 0, kk, lane), ldB_bfT(W1t, 512, kk, n0, lane), c);
    int n = n0 + (lane & 15), mB = (lane < 16) ? 0 : 8;
    float bv = b1[n];
#pragma unroll
    for (int r = 0; r < 8; ++r) {
      float v = c[r] + bv;
      sHid[(mB + r) * 256 + n] = (bf16_t)(v > 0.0f ? v : 0.0f);
    }
  }
  __syncthreads();
  {
    int n0 = wave * 16;
    v8f c = {};
#pragma unroll
    for (int kk = 0; kk < 256; kk += 32)
      c = wmma_bf16(ldA_bf(sHid, 256, 0, kk, lane), ldB_bfT(W2t, 256, kk, n0, lane), c);
    int n = n0 + (lane & 15), mB = (lane < 16) ? 0 : 8;
    float bv = b2[n];
#pragma unroll
    for (int r = 0; r < 8; ++r) {
      float v = c[r] + bv;
      size_t idx = (size_t)(row0 + mB + r) * kD + n;
      Hn[idx] = v;
      HBn[idx] = (bf16_t)v;
    }
  }
}

// fc head: t = relu(pad(h_side) @ fc1 + b1) @ fc2 + b2, row-masked -> MQ/MC (bf16)
__global__ void k_fc(const bf16_t* __restrict__ HB,
                     const bf16_t* __restrict__ W1t, const float* __restrict__ b1,
                     const bf16_t* __restrict__ W2t, const float* __restrict__ b2,
                     const int* __restrict__ qsz, const int* __restrict__ csz,
                     bf16_t* __restrict__ MQ, bf16_t* __restrict__ MC) {
  extern __shared__ bf16_t smh[];
  bf16_t* sx = smh;            // 128*128
  bf16_t* sh = smh + 16384;    // 128*128
  int bs = blockIdx.x, b = bs >> 1, s = bs & 1;
  const bf16_t* X = HB + (size_t)(2 * b + s) * kN * kD;
  int tid = threadIdx.x;
  for (int i = tid; i < 16384; i += 256) {
    int r = i >> 7;
    sx[i] = (r < kN) ? X[i] : (bf16_t)0.0f;
  }
  __syncthreads();
  int wave = tid >> 5, lane = tid & 31;
  for (int t = wave; t < 64; t += 8) {
    int m0 = (t >> 3) * 16, n0 = (t & 7) * 16;
    v8f c = {};
#pragma unroll
    for (int kk = 0; kk < 128; kk += 32)
      c = wmma_bf16(ldA_bf(sx, 128, m0, kk, lane), ldB_bfT(W1t, 128, kk, n0, lane), c);
    int n = n0 + (lane & 15), mB = (lane < 16) ? 0 : 8;
    float bv = b1[n];
#pragma unroll
    for (int r = 0; r < 8; ++r) {
      float v = c[r] + bv;
      sh[(m0 + mB + r) * 128 + n] = (bf16_t)(v > 0.0f ? v : 0.0f);
    }
  }
  __syncthreads();
  int size = s ? csz[b] : qsz[b];
  bf16_t* OUT = (s ? MC : MQ) + (size_t)b * 16384;
  for (int t = wave; t < 64; t += 8) {
    int m0 = (t >> 3) * 16, n0 = (t & 7) * 16;
    v8f c = {};
#pragma unroll
    for (int kk = 0; kk < 128; kk += 32)
      c = wmma_bf16(ldA_bf(sh, 128, m0, kk, lane), ldB_bfT(W2t, 128, kk, n0, lane), c);
    int n = n0 + (lane & 15), mB = (lane < 16) ? 0 : 8;
    float bv = b2[n];
#pragma unroll
    for (int r = 0; r < 8; ++r) {
      int m = m0 + mB + r;
      OUT[(size_t)m * 128 + n] = (bf16_t)((m < size) ? (c[r] + bv) : 0.0f);
    }
  }
}

// log_alpha = (mq @ mc^T) / SK_TEMP per batch; mq/mc bf16, K-contiguous rows
__global__ void k_logalpha(const bf16_t* __restrict__ MQ, const bf16_t* __restrict__ MC,
                           float* __restrict__ LA) {
  int b = blockIdx.x;
  const bf16_t* q  = MQ + (size_t)b * 16384;
  const bf16_t* cc = MC + (size_t)b * 16384;
  float* la = LA + (size_t)b * 16384;
  int tid = threadIdx.x, wave = tid >> 5, lane = tid & 31;
  for (int t = wave; t < 64; t += 8) {
    int m0 = (t >> 3) * 16, n0 = (t & 7) * 16;
    v8f c = {};
#pragma unroll
    for (int kk = 0; kk < 128; kk += 32)
      c = wmma_bf16(ldA_bf(q, 128, m0, kk, lane), ldB_bfT(cc, 128, kk, n0, lane), c);
    int n = n0 + (lane & 15), mB = (lane < 16) ? 0 : 8;
#pragma unroll
    for (int r = 0; r < 8; ++r) la[(size_t)(m0 + mB + r) * 128 + n] = c[r] * kInvTemp;
  }
}

// Sinkhorn (20 iters in LDS) + P = exp(la) + loss = -sum relu(sq - P@sc)
__global__ void k_sinkhorn(const float* __restrict__ LAg, const float* __restrict__ H,
                           float* __restrict__ out) {
  extern __shared__ float sm[];
  float* la  = sm;            // 16384
  float* red = sm + 16384;    // 256
  int b = blockIdx.x, tid = threadIdx.x;
  for (int i = tid; i < 16384; i += 256) la[i] = LAg[(size_t)b * 16384 + i];
  __syncthreads();
  for (int it = 0; it < kSKIT; ++it) {
    for (int r = tid; r < 128; r += 256) {          // axis=2 (rows)
      float mx = -3.4e38f;
      for (int j = 0; j < 128; ++j) mx = fmaxf(mx, la[r * 128 + j]);
      float s = 0.0f;
      for (int j = 0; j < 128; ++j) s += __expf(la[r * 128 + j] - mx);
      float lse = mx + __logf(s);
      for (int j = 0; j < 128; ++j) la[r * 128 + j] -= lse;
    }
    __syncthreads();
    for (int cI = tid; cI < 128; cI += 256) {       // axis=1 (cols)
      float mx = -3.4e38f;
      for (int i = 0; i < 128; ++i) mx = fmaxf(mx, la[i * 128 + cI]);
      float s = 0.0f;
      for (int i = 0; i < 128; ++i) s += __expf(la[i * 128 + cI] - mx);
      float lse = mx + __logf(s);
      for (int i = 0; i < 128; ++i) la[i * 128 + cI] -= lse;
    }
    __syncthreads();
  }
  for (int i = tid; i < 16384; i += 256) la[i] = __expf(la[i]);   // P
  __syncthreads();
  const float* sq = H + (size_t)(2 * b) * kN * kD;
  const float* sc = H + (size_t)(2 * b + 1) * kN * kD;
  int wave = tid >> 5, lane = tid & 31;
  float acc = 0.0f;
  for (int t = wave; t < 64; t += 8) {
    int m0 = (t >> 3) * 16, n0 = (t & 7) * 16;
    v8f c = {};
#pragma unroll
    for (int kk = 0; kk < 128; kk += 32)
      c = wmma_bf16(ldA_rm_f32(la, 128, m0, kk, lane),
                    ldB_rm_bound_f32(sc, kD, kk, n0, lane, kN), c);
    int n = n0 + (lane & 15), mB = (lane < 16) ? 0 : 8;
#pragma unroll
    for (int r = 0; r < 8; ++r) {
      int m = m0 + mB + r;
      float sqv = (m < kN) ? sq[(size_t)m * kD + n] : 0.0f;
      float v = sqv - c[r];
      acc += (v > 0.0f) ? v : 0.0f;
    }
  }
  red[tid] = acc;
  __syncthreads();
  for (int s = 128; s > 0; s >>= 1) {
    if (tid < s) red[tid] += red[tid + s];
    __syncthreads();
  }
  if (tid == 0) out[b] = -red[0];
}

// ---------------- host launcher ----------------
extern "C" void kernel_launch(void* const* d_in, const int* in_sizes, int n_in,
                              void* d_out, int out_size, void* d_ws, size_t ws_size,
                              hipStream_t stream) {
  (void)in_sizes; (void)n_in; (void)out_size; (void)ws_size;
  const float* nodef = (const float*)d_in[0];
  const float* edgef = (const float*)d_in[1];
  const int* from_idx = (const int*)d_in[2];
  const int* to_idx   = (const int*)d_in[3];
  const int* qsz = (const int*)d_in[5];
  const int* csz = (const int*)d_in[6];
  const float* encW = (const float*)d_in[7];
  const float* encB = (const float*)d_in[8];
  const float* eW  = (const float*)d_in[9];
  const float* eB  = (const float*)d_in[10];
  const float* mW1 = (const float*)d_in[11];
  const float* mb1 = (const float*)d_in[12];
  const float* mW2 = (const float*)d_in[13];
  const float* mb2 = (const float*)d_in[14];
  const float* uW1 = (const float*)d_in[15];
  const float* ub1 = (const float*)d_in[16];
  const float* uW2 = (const float*)d_in[17];
  const float* ub2 = (const float*)d_in[18];
  const float* f1W = (const float*)d_in[19];
  const float* f1b = (const float*)d_in[20];
  const float* f2W = (const float*)d_in[21];
  const float* f2b = (const float*)d_in[22];

  // ---- f32 workspace ----
  float* ws = (float*)d_ws;
  float* hA  = ws;
  float* hB  = hA  + (size_t)kNODES * kD;
  float* AGG = hB  + (size_t)kNODES * kD;
  float* CR  = AGG + (size_t)kNODES * 256;
  float* LA  = CR  + (size_t)kNODES * kD;
  // ---- bf16 workspace (16B-aligned: f32 region is a multiple of 16 bytes) ----
  bf16_t* bws = (bf16_t*)(LA + (size_t)kB * kMAXN * kMAXN);
  bf16_t* EB   = bws;                           // edges bf16: kEDGES x 128
  bf16_t* hbA  = EB  + (size_t)kEDGES * kD;     // h shadow A
  bf16_t* hbB  = hbA + (size_t)kNODES * kD;     // h shadow B
  bf16_t* wtEncN = hbB + (size_t)kNODES * kD;   // 128x128 (N x K)
  bf16_t* wtEncE = wtEncN + 128 * 128;          // 128x32
  bf16_t* wtM1   = wtEncE + 128 * 32;           // 256x384
  bf16_t* wtM2   = wtM1 + 256 * 384;            // 256x256
  bf16_t* wtU1   = wtM2 + 256 * 256;            // 256x512
  bf16_t* wtU2   = wtU1 + 256 * 512;            // 128x256
  bf16_t* wtF1   = wtU2 + 128 * 256;            // 128x128
  bf16_t* wtF2   = wtF1 + 128 * 128;            // 128x128
  bf16_t* MQ     = wtF2 + 128 * 128;            // B x 128 x 128
  bf16_t* MC     = MQ + (size_t)kB * kMAXN * kMAXN;

  constexpr size_t crossLds = 96 * 96 * sizeof(float) +
                              (2 * 96 * 128 + 2 * 128 * 96 + 2 * 96 * 96) * sizeof(bf16_t);
  constexpr size_t fcLds    = 2 * 16384 * sizeof(bf16_t);           // 64KB
  constexpr size_t skLds    = (16384 + 256) * sizeof(float);        // 66560
  hipFuncSetAttribute((const void*)k_cross,    hipFuncAttributeMaxDynamicSharedMemorySize, (int)crossLds);
  hipFuncSetAttribute((const void*)k_fc,       hipFuncAttributeMaxDynamicSharedMemorySize, (int)fcLds);
  hipFuncSetAttribute((const void*)k_sinkhorn, hipFuncAttributeMaxDynamicSharedMemorySize, (int)skLds);

  // weight prep: transpose + bf16-convert (tiny)
  k_wprep<<<(128 * 128 + 255) / 256, 256, 0, stream>>>(encW, wtEncN, 128, 128);
  k_wprep<<<(32 * 128 + 255) / 256, 256, 0, stream>>>(eW, wtEncE, 32, 128);
  k_wprep<<<(384 * 256 + 255) / 256, 256, 0, stream>>>(mW1, wtM1, 384, 256);
  k_wprep<<<(256 * 256 + 255) / 256, 256, 0, stream>>>(mW2, wtM2, 256, 256);
  k_wprep<<<(512 * 256 + 255) / 256, 256, 0, stream>>>(uW1, wtU1, 512, 256);
  k_wprep<<<(256 * 128 + 255) / 256, 256, 0, stream>>>(uW2, wtU2, 256, 128);
  k_wprep<<<(128 * 128 + 255) / 256, 256, 0, stream>>>(f1W, wtF1, 128, 128);
  k_wprep<<<(128 * 128 + 255) / 256, 256, 0, stream>>>(f2W, wtF2, 128, 128);

  k_encode_nodes<<<kNODES / 16, 256, 0, stream>>>(nodef, wtEncN, encB, hA, hbA);
  k_encode_edges<<<kEDGES / 16, 256, 0, stream>>>(edgef, wtEncE, eB, EB);

  float* cur = hA;   float* nxt = hB;
  bf16_t* curB = hbA; bf16_t* nxtB = hbB;
  for (int it = 0; it < 5; ++it) {
    k_zero<<<2048, 256, 0, stream>>>(AGG, (size_t)kNODES * 256);
    k_msg<<<kEDGES / 16, 256, 0, stream>>>(curB, EB, from_idx, to_idx, wtM1, mb1, wtM2, mb2, AGG);
    k_cross<<<kB, 256, crossLds, stream>>>(cur, curB, CR);
    k_update<<<kNODES / 16, 256, 0, stream>>>(curB, AGG, CR, wtU1, ub1, wtU2, ub2, nxt, nxtB);
    float* t = cur; cur = nxt; nxt = t;
    bf16_t* tb = curB; curB = nxtB; nxtB = tb;
  }

  k_fc<<<2 * kB, 256, fcLds, stream>>>(curB, wtF1, f1b, wtF2, f2b, qsz, csz, MQ, MC);
  k_logalpha<<<kB, 256, 0, stream>>>(MQ, MC, LA);
  k_sinkhorn<<<kB, 256, skLds, stream>>>(LA, cur, (float*)d_out);
}
